// KNRM_74706661146685
// MI455X (gfx1250) — matrix operand order repaired and checked by reading, not compiled, not run on hardware
//
#include <hip/hip_runtime.h>

// KNRM ranking kernel for gfx1250 (MI455X), wave32 + WMMA.
// B=128 batches, Q=16 queries, L=2048 docs, E=128 emb dim, K=11 RBF kernels.

typedef __attribute__((ext_vector_type(16))) _Float16 v16h;
typedef __attribute__((ext_vector_type(8)))  float    v8f;

#define NQ     16
#define NL     2048
#define NE     128
#define NK     11
#define WAVES  8
#define NTILES (NL / 16)   // 128 doc tiles of 16
#define SIGInv 50.0f       // 1/(2*0.1^2)
#define EPS    1e-8f

// ---- wave32 cross-lane helpers (raw builtins; no wave64 assumptions) ----
template <int MASK>
__device__ __forceinline__ float swz_xor_add(float v) {
    // ds_swizzle group-of-32: offset = {0, xor[14:10], or[9:5], and[4:0]}
    const int imm = (MASK << 10) | 0x1f;
    int o = __builtin_amdgcn_ds_swizzle(__float_as_int(v), imm);
    return v + __int_as_float(o);
}

__device__ __forceinline__ float lane_gather(float v, int srcLane) {
    return __int_as_float(
        __builtin_amdgcn_ds_bpermute(srcLane << 2, __float_as_int(v)));
}

// convert 4 f32 into v16h[off..off+3], accumulate sum of squares
__device__ __forceinline__ void cvt4(v16h& dst, int off, float4 f, float& ss) {
    ss += f.x * f.x + f.y * f.y + f.z * f.z + f.w * f.w;
    dst[off + 0] = (_Float16)f.x;
    dst[off + 1] = (_Float16)f.y;
    dst[off + 2] = (_Float16)f.z;
    dst[off + 3] = (_Float16)f.w;
}

__device__ __forceinline__ float4 ldg4(const float* p) {
    return *(const float4*)p;
}

__global__ __launch_bounds__(256) void knrm_kernel(
    const int*   __restrict__ queries,    // (B,16)
    const int*   __restrict__ documents,  // (B,2048)
    const float* __restrict__ emb,        // (100000,128)
    const float* __restrict__ W,          // (11)
    const float* __restrict__ bias,       // (1)
    float*       __restrict__ out,        // (B)
    int B)
{
    __shared__ float pooled_s[NQ * NK];
    __shared__ float red_s[NK];

    const int batch  = blockIdx.x;
    const int tid    = threadIdx.x;
    const int wave   = tid >> 5;
    const int lane   = tid & 31;
    const int half   = lane >> 4;   // which 16-lane half
    const int lane16 = lane & 15;

    for (int i = tid; i < NQ * NK; i += blockDim.x) pooled_s[i] = 0.0f;
    __syncthreads();

    // ---------------- A fragments: query embeddings (16 x 128), f32 -> f16 ----
// Contiguous-k 16-bit A layout used here (16x32): lane<16 -> row=lane; elems 0..15 hold K=16h+0..15
// (h = lane/16), matching B. Four K-steps of 32.
    const int    qidx = queries[batch * NQ + lane16];
    const float* qp   = emb + (long)qidx * NE;

    v16h  afrag[4];
    float qss = 0.0f;
#pragma unroll
    for (int t = 0; t < 4; ++t) {
        const int base = 32 * t + 16 * half;
        cvt4(afrag[t], 0,  ldg4(qp + base +  0), qss);
        cvt4(afrag[t], 4,  ldg4(qp + base +  4), qss);
        cvt4(afrag[t], 8,  ldg4(qp + base +  8), qss);
        cvt4(afrag[t], 12, ldg4(qp + base + 12), qss);
    }
    qss = swz_xor_add<16>(qss);          // combine the two K-halves of the row
    const float qn = sqrtf(qss);         // ||q[lane16]|| (identical in both halves)

    // per-lane copies of qn for the 8 C-matrix rows this lane owns (m = 8h + r)
    float qnv[8];
#pragma unroll
    for (int r = 0; r < 8; ++r)
        qnv[r] = lane_gather(qn, 8 * half + r);

    // register RBF accumulators: acc[r][k] for rows m=8h+r, summed over this
    // lane's doc columns across all of its tiles
    float acc[8][NK];
#pragma unroll
    for (int r = 0; r < 8; ++r)
#pragma unroll
        for (int k = 0; k < NK; ++k) acc[r][k] = 0.0f;

    const int* didx_p = documents + batch * NL;

    // ---------------- main loop: each wave owns tiles wave, wave+8, ... ----
    for (int tile = wave; tile < NTILES; tile += WAVES) {
        // B fragments: 16-bit B layout (32x16): lane<16 -> col=lane,
        // elems 0..15 hold K = 16h + 0..15 per 32-K step.
        const int    j    = tile * 16 + lane16;
        const float* dp   = emb + (long)didx_p[j] * NE;

        v16h  bfrag[4];
        float dss = 0.0f;
#pragma unroll
        for (int t = 0; t < 4; ++t) {
            const int base = 32 * t + 16 * half;
            cvt4(bfrag[t], 0,  ldg4(dp + base +  0), dss);
            cvt4(bfrag[t], 4,  ldg4(dp + base +  4), dss);
            cvt4(bfrag[t], 8,  ldg4(dp + base +  8), dss);
            cvt4(bfrag[t], 12, ldg4(dp + base + 12), dss);
        }
        dss = swz_xor_add<16>(dss);
        const float dn = sqrtf(dss);     // ||d[tile*16 + lane16]||

        // 16x16 dot tile via 4 chained WMMAs over K=128
        v8f c = {};
#pragma unroll
        for (int t = 0; t < 4; ++t)
            c = __builtin_amdgcn_wmma_f32_16x16x32_f16(
                    false, afrag[t], false, bfrag[t],
                    (short)0, c, false, false);

        // C layout: c[r] = dot(q[8h+r], d[tile*16+lane16])
#pragma unroll
        for (int r = 0; r < 8; ++r) {
            const float sim = c[r] / (qnv[r] * dn + EPS);
#pragma unroll
            for (int k = 0; k < NK; ++k) {
                const float mu  = -1.0f + 0.2f * (float)k;
                const float dlt = sim - mu;
                acc[r][k] += __expf(-dlt * dlt * SIGInv);
            }
        }
    }

    // ---------------- reduce doc-columns: XOR butterfly within each half ----
#pragma unroll
    for (int r = 0; r < 8; ++r)
#pragma unroll
        for (int k = 0; k < NK; ++k) {
            float v = acc[r][k];
            v = swz_xor_add<1>(v);
            v = swz_xor_add<2>(v);
            v = swz_xor_add<4>(v);
            v = swz_xor_add<8>(v);
            acc[r][k] = v;
        }

    if (lane16 == 0) {   // lanes 0 (rows 0..7) and 16 (rows 8..15)
        const int m0 = 8 * half;
#pragma unroll
        for (int r = 0; r < 8; ++r)
#pragma unroll
            for (int k = 0; k < NK; ++k)
                atomicAdd(&pooled_s[(m0 + r) * NK + k], acc[r][k]);
    }
    __syncthreads();

    // ---------------- epilogue: log-pool over queries, linear, sigmoid ----
    if (tid < NK) {
        float st = 0.0f;
        for (int m = 0; m < NQ; ++m) st += __logf(pooled_s[m * NK + tid]);
        red_s[tid] = st * W[tid];
    }
    __syncthreads();
    if (tid == 0) {
        float z = bias[0];
#pragma unroll
        for (int k = 0; k < NK; ++k) z += red_s[k];
        out[batch] = 1.0f / (1.0f + __expf(-z));
    }
}

extern "C" void kernel_launch(void* const* d_in, const int* in_sizes, int n_in,
                              void* d_out, int out_size, void* d_ws, size_t ws_size,
                              hipStream_t stream) {
    const int*   queries   = (const int*)d_in[0];
    const int*   documents = (const int*)d_in[1];
    const float* emb       = (const float*)d_in[2];
    const float* W         = (const float*)d_in[3];
    const float* bias      = (const float*)d_in[4];
    float*       out       = (float*)d_out;

    const int B = in_sizes[0] / NQ;   // 128
    hipLaunchKernelGGL(knrm_kernel, dim3(B), dim3(256), 0, stream,
                       queries, documents, emb, W, bias, out, B);
}
